// Head_3015067041898
// MI455X (gfx1250) — matrix operand order repaired and checked by reading, not compiled, not run on hardware
//
#include <hip/hip_runtime.h>
#include <hip/hip_bf16.h>
#include <math.h>

typedef __attribute__((ext_vector_type(16))) __bf16 bf16x16;
typedef __attribute__((ext_vector_type(8)))  __bf16 bf16x8;
typedef __attribute__((ext_vector_type(8)))  float  v8f;
typedef unsigned int __attribute__((ext_vector_type(4))) u32x4;
typedef int __attribute__((ext_vector_type(4))) i32x4;
typedef int __attribute__((ext_vector_type(8))) i32x8;

#define B_  8
#define T_  2048
#define C_  1024
#define HS_ 128
#define M_  (B_ * T_)

#ifndef __has_builtin
#define __has_builtin(x) 0
#endif

// TDM builtins only matter (and only exist) on the device pass.
#if defined(__HIP_DEVICE_COMPILE__)
#if !(__has_builtin(__builtin_amdgcn_tensor_load_to_lds) && \
      __has_builtin(__builtin_amdgcn_s_wait_tensorcnt))
#error "gfx1250 TDM builtins not available on this toolchain (device pass)"
#endif
#define USE_TDM 1
#else
#define USE_TDM 0   // host pass: parse-only stubs
#endif

// LDS layout of attention kernel (dynamic LDS only => region base 0)
// double-buffered K / V^T tiles, each 128x136 bf16
#define KB0_OFF  0u
#define VT0_OFF  34816u
#define KB1_OFF  69632u
#define VT1_OFF  104448u
#define SP_OFF   139264u                // 8 waves * 16 * 136 bf16
#define SMEM_SZ  174080u

// ---------- helpers ----------

__device__ __forceinline__ __bf16 f2bf(float f) {
  unsigned u = __builtin_bit_cast(unsigned, f);
  unsigned r = u + 0x7FFFu + ((u >> 16) & 1u);   // round-to-nearest-even
  unsigned short h = (unsigned short)(r >> 16);
  return __builtin_bit_cast(__bf16, h);
}

// A-matrix 16x32 bf16 fragment from row-major [rows][lda] storage.
__device__ __forceinline__ bf16x16 load_frag_a(const __bf16* base, int row0,
                                               int lda, int kc, int lane) {
  int half = lane >> 4, r = lane & 15;
  const __bf16* p = base + (size_t)(row0 + r) * lda + kc + half * 8;
  union { bf16x16 v; bf16x8 h[2]; } u;
  u.h[0] = *(const bf16x8*)(p);
  u.h[1] = *(const bf16x8*)(p + 16);
  return u.v;
}

// B-matrix 32x16 bf16 fragment from n-major (transposed) [n][lda] storage.
__device__ __forceinline__ bf16x16 load_frag_b(const __bf16* base, int row0,
                                               int lda, int kc, int lane) {
  int half = lane >> 4, r = lane & 15;
  const __bf16* p = base + (size_t)(row0 + r) * lda + kc + half * 8;
  union { bf16x16 v; bf16x8 h[2]; } u;
  u.h[0] = *(const bf16x8*)(p);
  u.h[1] = *(const bf16x8*)(p + 16);
  return u.v;
}

__device__ __forceinline__ v8f wmma_bf16(bf16x16 a, bf16x16 b, v8f c) {
  return __builtin_amdgcn_wmma_f32_16x16x32_bf16(
      false, a, false, b, (short)0, c, false, false);
}

// TDM: DMA a 128x128 bf16 tile (row stride = stride_elems) into LDS at byte
// offset lds_off, padded +16B per 256B row => lda 136 elements.
__device__ __forceinline__ void tdm_load_tile_128x128(const void* gaddr,
                                                      unsigned lds_off,
                                                      int stride_elems) {
#if USE_TDM
  unsigned long long ga = (unsigned long long)(uintptr_t)gaddr;
  u32x4 g0;
  g0[0] = 1u;                                   // count=1 (valid), user mode
  g0[1] = lds_off;                              // LDS byte address
  g0[2] = (unsigned)(ga & 0xFFFFFFFFull);       // global_addr[31:0]
  g0[3] = ((unsigned)(ga >> 32) & 0x01FFFFFFu)  // global_addr[56:32]
          | 0x80000000u;                        // type=2 ("image")
  i32x8 g1;
  g1[0] = (1 << 16)                             // data_size = 2 bytes
        | (1 << 20)                             // pad_enable
        | (5 << 22)                             // pad_interval 2^(5+1)=64dw=256B
        | (3 << 25);                            // pad_amount 3+1 = 4dw = 16B
  g1[1] = 128 << 16;                            // tensor_dim0 = 128
  g1[2] = 128 << 16;                            // tensor_dim1 = 128
  g1[3] = 128 << 16;                            // tile_dim0 = 128
  g1[4] = 128;                                  // tile_dim1 = 128
  g1[5] = stride_elems;                         // tensor_dim0_stride
  g1[6] = 0;
  g1[7] = 0;
  i32x4 z4 = {0, 0, 0, 0};
#if defined(__clang_major__) && (__clang_major__ >= 23)
  i32x8 z8 = {0, 0, 0, 0, 0, 0, 0, 0};
  __builtin_amdgcn_tensor_load_to_lds(g0, g1, z4, z4, z8, 0);
#else
  __builtin_amdgcn_tensor_load_to_lds(g0, g1, z4, z4, 0);
#endif
#else
  (void)gaddr; (void)lds_off; (void)stride_elems;
#endif
}

__device__ __forceinline__ void wait_tensorcnt0() {
#if USE_TDM
  __builtin_amdgcn_s_wait_tensorcnt(0);
#endif
}

// ---------- kernel 0: RoPE cos/sin tables (matches reference formula) ----------

__global__ __launch_bounds__(256) void rope_table_kernel(float* __restrict__ Ct,
                                                         float* __restrict__ St) {
  int idx = blockIdx.x * 256 + threadIdx.x;   // T_ * 64 entries
  int t = idx >> 6, i = idx & 63;
  float theta = __powf(10000.0f, -2.0f * ((float)i - 1.0f) / 64.0f);
  float sn, cs;
  __sincosf((float)t * theta, &sn, &cs);
  Ct[idx] = cs;
  St[idx] = sn;
}

// ---------- kernel 1: q/k/v projection + RoPE, fp32 -> bf16 ----------
// grid (M_/128, 3), block 256. Q,K: [m][128]; V transposed: Vt[b][hs][t].

__global__ __launch_bounds__(256) void proj_rope_kernel(
    const float* __restrict__ x,
    const float* __restrict__ Wq, const float* __restrict__ Wk,
    const float* __restrict__ Wv,
    const float* __restrict__ Ct, const float* __restrict__ St,
    __bf16* __restrict__ Qw, __bf16* __restrict__ Kw, __bf16* __restrict__ Vt) {
  __shared__ __bf16 sX[128 * 48];
  __shared__ __bf16 sWt[128 * 48];

  const int j    = blockIdx.y;
  const float* W = (j == 0) ? Wq : (j == 1) ? Wk : Wv;

  const int m0   = blockIdx.x * 128;
  const int tid  = threadIdx.x;
  const int lane = tid & 31;
  const int w    = tid >> 5;

  v8f acc[8];
#pragma unroll
  for (int nt = 0; nt < 8; ++nt)
#pragma unroll
    for (int c = 0; c < 8; ++c) acc[nt][c] = 0.0f;

  for (int kc = 0; kc < C_; kc += 32) {
    {  // stage x tile 128x32 -> bf16
      int r  = tid >> 1;
      int c0 = (tid & 1) * 16;
      const float* src = x + (size_t)(m0 + r) * C_ + kc + c0;
      __bf16* dst = sX + r * 48 + c0;
#pragma unroll
      for (int e = 0; e < 16; ++e) dst[e] = f2bf(src[e]);
    }
    {  // stage W^T tile: sWt[col][k]
      int kk = tid >> 3;
      int c0 = (tid & 7) * 16;
      const float* src = W + (size_t)(kc + kk) * HS_ + c0;
#pragma unroll
      for (int e = 0; e < 16; ++e) sWt[(c0 + e) * 48 + kk] = f2bf(src[e]);
    }
    __syncthreads();

    bf16x16 a = load_frag_a(sX, 16 * w, 48, 0, lane);
    bf16x16 bm[8];
#pragma unroll
    for (int nt = 0; nt < 8; ++nt) bm[nt] = load_frag_b(sWt, 16 * nt, 48, 0, lane);
#pragma unroll
    for (int nt = 0; nt < 8; ++nt) acc[nt] = wmma_bf16(a, bm[nt], acc[nt]);
    __syncthreads();
  }

  // RoPE (q,k) via precomputed tables + store
  const int  half = lane >> 4, lc = lane & 15;
  const bool do_rope = (j < 2);
#pragma unroll
  for (int nt = 0; nt < 8; ++nt) {
    int n = nt * 16 + lc;
    int i = n >> 1;
    bool odd = (n & 1) != 0;
#pragma unroll
    for (int v = 0; v < 8; ++v) {
      int   m   = m0 + 16 * w + v + 8 * half;
      float val = acc[nt][v];
      if (do_rope) {
        int   ti = (m & (T_ - 1)) * 64 + i;
        float cs = Ct[ti], sn = St[ti];
        float p = __shfl_xor(val, 1, 32);
        val = odd ? (-p * sn + val * cs) : (val * cs + p * sn);
      }
      if (j == 2) {  // V transposed: Vt[b][n][t]
        Vt[((size_t)(m >> 11) * HS_ + n) * T_ + (m & (T_ - 1))] = f2bf(val);
      } else {
        __bf16* Dst = (j == 0) ? Qw : Kw;
        Dst[(size_t)m * HS_ + n] = f2bf(val);
      }
    }
  }
}

// ---------- kernel 2: causal flash attention, double-buffered TDM ----------
// grid (T_/128, B_), block 256 (8 waves, each wave owns 16 query rows).

__global__ __launch_bounds__(256) void attn_kernel(
    const __bf16* __restrict__ Qw, const __bf16* __restrict__ Kw,
    const __bf16* __restrict__ Vt, float* __restrict__ out) {
  extern __shared__ __align__(16) char smem[];
  __bf16* sP = (__bf16*)(smem + SP_OFF);

  const int b    = blockIdx.y;
  const int qb   = blockIdx.x;
  const int tid  = threadIdx.x;
  const int lane = tid & 31;
  const int w    = tid >> 5;
  const int half = lane >> 4, lc = lane & 15;

  const size_t rowbase = (size_t)b * T_;
  const int q0 = qb * 128;
  const int qw = q0 + 16 * w;

  bf16x16 qf[4];
#pragma unroll
  for (int kc = 0; kc < 4; ++kc)
    qf[kc] = load_frag_a(Qw + rowbase * HS_, qw, HS_, kc * 32, lane);

  v8f   O[8];
  float m_r[8], l_r[8];
#pragma unroll
  for (int ht = 0; ht < 8; ++ht)
#pragma unroll
    for (int c = 0; c < 8; ++c) O[ht][c] = 0.0f;
#pragma unroll
  for (int v = 0; v < 8; ++v) { m_r[v] = -__builtin_inff(); l_r[v] = 0.0f; }

  const float rs = 0.08838834764831845f;  // 1/sqrt(128)

  // prologue: kick off DMA for block 0 into buffer 0
  if (w == 0) {
    tdm_load_tile_128x128(Kw + rowbase * HS_, KB0_OFF, HS_);
    tdm_load_tile_128x128(Vt + (size_t)b * HS_ * T_, VT0_OFF, T_);
  }

  for (int jb = 0; jb <= qb; ++jb) {
    const int cur = jb & 1;
    if (w == 0) wait_tensorcnt0();   // buf[cur] landed
    __syncthreads();                 // publish to all waves

    // prefetch next block into the other buffer (overlaps with compute below)
    if (w == 0 && jb < qb) {
      const unsigned kOffN  = cur ? KB0_OFF : KB1_OFF;
      const unsigned vtOffN = cur ? VT0_OFF : VT1_OFF;
      tdm_load_tile_128x128(Kw + (rowbase + (size_t)(jb + 1) * 128) * HS_,
                            kOffN, HS_);
      tdm_load_tile_128x128(Vt + (size_t)b * HS_ * T_ + (size_t)(jb + 1) * 128,
                            vtOffN, T_);
    }

    // select current buffer via integer offsets (no LDS pointer arrays)
    const __bf16* sK  = (const __bf16*)(smem + (cur ? KB1_OFF : KB0_OFF));
    const __bf16* sVT = (const __bf16*)(smem + (cur ? VT1_OFF : VT0_OFF));

    // S = Q * K^T : per K-chunk load all 8 B-frags, then 8 independent WMMAs
    v8f S[8];
#pragma unroll
    for (int nt = 0; nt < 8; ++nt)
#pragma unroll
      for (int c = 0; c < 8; ++c) S[nt][c] = 0.0f;
#pragma unroll
    for (int kc = 0; kc < 4; ++kc) {
      bf16x16 kb[8];
#pragma unroll
      for (int nt = 0; nt < 8; ++nt)
        kb[nt] = load_frag_b(sK, nt * 16, 136, kc * 32, lane);
#pragma unroll
      for (int nt = 0; nt < 8; ++nt)
        S[nt] = wmma_bf16(qf[kc], kb[nt], S[nt]);
    }

    // scale + causal mask
#pragma unroll
    for (int nt = 0; nt < 8; ++nt) {
      int kg = jb * 128 + nt * 16 + lc;
#pragma unroll
      for (int v = 0; v < 8; ++v) {
        int qg = qw + v + 8 * half;
        float sc = S[nt][v] * rs;
        S[nt][v] = (kg > qg) ? -__builtin_inff() : sc;
      }
    }

    // online softmax (half-wave shfl_xor reductions keep row groups separate)
#pragma unroll
    for (int v = 0; v < 8; ++v) {
      float mx = m_r[v];
#pragma unroll
      for (int nt = 0; nt < 8; ++nt) mx = fmaxf(mx, S[nt][v]);
#pragma unroll
      for (int off = 8; off >= 1; off >>= 1)
        mx = fmaxf(mx, __shfl_xor(mx, off, 32));
      float alpha = __expf(m_r[v] - mx);
      float ls = 0.0f;
#pragma unroll
      for (int nt = 0; nt < 8; ++nt) {
        float p = __expf(S[nt][v] - mx);
        S[nt][v] = p;
        ls += p;
      }
#pragma unroll
      for (int off = 8; off >= 1; off >>= 1)
        ls += __shfl_xor(ls, off, 32);
      l_r[v] = l_r[v] * alpha + ls;
      m_r[v] = mx;
#pragma unroll
      for (int ht = 0; ht < 8; ++ht) O[ht][v] *= alpha;
    }

    // C-layout -> A-layout reshape of P through per-wave LDS strip
    __bf16* pW = sP + w * 16 * 136;
#pragma unroll
    for (int nt = 0; nt < 8; ++nt)
#pragma unroll
      for (int v = 0; v < 8; ++v)
        pW[(v + 8 * half) * 136 + nt * 16 + lc] = f2bf(S[nt][v]);

    // O += P * V : per key-chunk load P frag + all 8 V^T frags, then 8 WMMAs
#pragma unroll
    for (int kc2 = 0; kc2 < 4; ++kc2) {
      bf16x16 pf = load_frag_a(pW, 0, 136, kc2 * 32, lane);
      bf16x16 vf[8];
#pragma unroll
      for (int ht = 0; ht < 8; ++ht)
        vf[ht] = load_frag_b(sVT, ht * 16, 136, kc2 * 32, lane);
#pragma unroll
      for (int ht = 0; ht < 8; ++ht)
        O[ht] = wmma_bf16(pf, vf[ht], O[ht]);
    }
    __syncthreads();  // all waves done with buf[cur] before it is re-targeted
  }

  // epilogue: normalize and store fp32
#pragma unroll
  for (int v = 0; v < 8; ++v) {
    float inv = 1.0f / l_r[v];
#pragma unroll
    for (int ht = 0; ht < 8; ++ht) {
      int qg = qw + v + 8 * half;
      out[(rowbase + qg) * HS_ + ht * 16 + lc] = O[ht][v] * inv;
    }
  }
}

// ---------- launch ----------

extern "C" void kernel_launch(void* const* d_in, const int* in_sizes, int n_in,
                              void* d_out, int out_size, void* d_ws, size_t ws_size,
                              hipStream_t stream) {
  const float* x  = (const float*)d_in[0];
  const float* Wq = (const float*)d_in[1];
  const float* Wk = (const float*)d_in[2];
  const float* Wv = (const float*)d_in[3];
  float* out = (float*)d_out;

  __bf16* Qw = (__bf16*)d_ws;
  __bf16* Kw = Qw + (size_t)M_ * HS_;
  __bf16* Vt = Kw + (size_t)M_ * HS_;              // Vt[b][hs][t]
  float*  Ct = (float*)(Vt + (size_t)M_ * HS_);    // T_ x 64
  float*  St = Ct + (size_t)T_ * 64;

  rope_table_kernel<<<dim3(T_ * 64 / 256), 256, 0, stream>>>(Ct, St);
  proj_rope_kernel<<<dim3(M_ / 128, 3), 256, 0, stream>>>(x, Wq, Wk, Wv, Ct, St,
                                                          Qw, Kw, Vt);
  attn_kernel<<<dim3(T_ / 128, B_), 256, SMEM_SZ, stream>>>(Qw, Kw, Vt, out);
}
